// DAL_50663434224529
// MI455X (gfx1250) — matrix operand order repaired and checked by reading, not compiled, not run on hardware
//
#include <hip/hip_runtime.h>
#include <math.h>

// ---------------------------------------------------------------------------
// dilated 3-tap per-channel attention:  q,k,v = 1x1 conv; softmax over taps
// B=16, C=512, T=2048, O=512, d=4
//  Phase 0: fp32 -> f16 conversion of x and weights (streaming)
//  Phase 1: fused QKV GEMM, v_wmma_f32_16x16x32_f16; weight tiles staged in
//           LDS via global_load_async_to_lds_b128 (ASYNCcnt); x tile staged
//           transposed with packed ds_store_b128. Block tile 64(M) x 64(N),
//           wave tile 16(M) x 32(N) x 3 weights => 6 WMMA per wave per K-step.
//  Phase 2: 3-tap softmax epilogue (zero padding => boundary taps exactly 0)
// ---------------------------------------------------------------------------

typedef __attribute__((ext_vector_type(16))) _Float16 v16h;
typedef __attribute__((ext_vector_type(8)))  _Float16 v8h;
typedef __attribute__((ext_vector_type(4)))  _Float16 v4h;
typedef __attribute__((ext_vector_type(8)))  float    v8f;
typedef __attribute__((ext_vector_type(4)))  float    v4f;

#define BSZ   16
#define CDIM  512
#define TDIM  2048
#define ODIM  512
#define DIL   4

#define TILE_M 64   // O rows per block
#define TILE_N 64   // T cols per block
#define TILE_K 32   // C depth per iteration
#define LDSP   40   // padded LDS row (halves): 80B rows, 16B aligned,
                    // 20-dword stride -> conflict-free 16-lane reads

// ---------------------------------------------------------------------------
__global__ __launch_bounds__(256) void cvt_f32_f16(const float* __restrict__ in,
                                                   _Float16* __restrict__ out)
{
    const size_t i = ((size_t)blockIdx.x * 256 + threadIdx.x) * 4;
    const v4f f = *(const v4f*)(in + i);
    v4h h;
    h.x = (_Float16)f.x; h.y = (_Float16)f.y;
    h.z = (_Float16)f.z; h.w = (_Float16)f.w;
    *(v4h*)(out + i) = h;
}

// ---------------------------------------------------------------------------
__device__ __forceinline__ void async_copy_b128(unsigned lds_off, const void* gptr)
{
    // per-lane 16B copy, global -> LDS, tracked with ASYNCcnt
    asm volatile("global_load_async_to_lds_b128 %0, %1, off"
                 :: "v"(lds_off), "v"((unsigned long long)(uintptr_t)gptr)
                 : "memory");
}

__global__ __launch_bounds__(256) void qkv_gemm_f16_wmma(
    const _Float16* __restrict__ xh,   // [B][C][T] f16
    const _Float16* __restrict__ wh,   // [3][O][C] f16
    _Float16* __restrict__ qh,
    _Float16* __restrict__ kh,
    _Float16* __restrict__ vh)
{
    __shared__ __align__(16) _Float16 Ah[3][TILE_M][LDSP]; // weight tiles
    __shared__ __align__(16) _Float16 Bh[TILE_N][LDSP];    // x tile, transposed [t][c]

    const int tid  = threadIdx.x;
    const int tt   = blockIdx.x;   // T tile (64 cols)
    const int ot   = blockIdx.y;   // O tile (64 rows)
    const int b    = blockIdx.z;

    const int wave = tid >> 5;
    const int lane = tid & 31;
    const int half = lane >> 4;    // 0: lanes 0-15, 1: lanes 16-31
    const int l16  = lane & 15;
    const int mi   = wave & 3;     // M sub-tile 0..3 (16 rows)
    const int ni   = wave >> 2;    // N pair 0..1 (32 cols each)

    // LDS byte offsets (low 32 bits of flat shared address == LDS offset)
    const unsigned AhBase = (unsigned)(uintptr_t)(&Ah[0][0][0]);

    // weight async copy: 3 weights x 64 rows x 4 chunks = 768 x 16B chunks,
    // 256 threads -> 3 chunks each; chunk group i == weight index i.
    const int wr  = tid >> 2;          // row 0..63
    const int wc8 = (tid & 3) * 8;     // col chunk 0,8,16,24

    // x tile staging: 32(K) x 64(T) halves = 2048; 8 per thread.
    // thread handles column xtl, channels cq*8 .. cq*8+7  -> one b128 store
    const int xtl = tid & 63;          // 0..63
    const int xcq = tid >> 6;          // 0..3

    v8f acc[3][2] = {};

    for (int c0 = 0; c0 < CDIM; c0 += TILE_K) {
        // ---- async weight tiles: pure f16 copies, global -> LDS ----
#pragma unroll
        for (int i = 0; i < 3; ++i) {
            const _Float16* g = wh + (((size_t)i * ODIM + ot * TILE_M + wr) * CDIM
                                      + c0 + wc8);
            const unsigned d = AhBase + (unsigned)(((i * TILE_M + wr) * LDSP + wc8) * 2);
            async_copy_b128(d, g);
        }
        // ---- x tile, transposed: 8 coalesced u16 loads + one ds_store_b128 ----
        {
            const unsigned short* xs = (const unsigned short*)xh
                + ((size_t)b * CDIM + c0 + xcq * 8) * TDIM
                + (size_t)tt * TILE_N + xtl;
            uint4 pk;
            {
                const unsigned u0 = xs[0];
                const unsigned u1 = xs[TDIM];
                const unsigned u2 = xs[2 * TDIM];
                const unsigned u3 = xs[3 * TDIM];
                const unsigned u4 = xs[4 * TDIM];
                const unsigned u5 = xs[5 * TDIM];
                const unsigned u6 = xs[6 * TDIM];
                const unsigned u7 = xs[7 * TDIM];
                pk.x = u0 | (u1 << 16);
                pk.y = u2 | (u3 << 16);
                pk.z = u4 | (u5 << 16);
                pk.w = u6 | (u7 << 16);
            }
            *(uint4*)(&Bh[xtl][xcq * 8]) = pk;
        }
        asm volatile("s_wait_asynccnt 0x0" ::: "memory");
        __syncthreads();

        // ---- B fragments: columns N = ni*32 + n*16 + l16, K 8*half..+7, 16+8*half..+7
        v16h bfrag[2];
#pragma unroll
        for (int n = 0; n < 2; ++n) {
            const v8h* browp = (const v8h*)(&Bh[ni * 32 + n * 16 + l16][0]);
            const v8h bc0 = browp[half];
            const v8h bc1 = browp[half + 2];
            bfrag[n] = __builtin_shufflevector(
                bc0, bc1, 0, 1, 2, 3, 4, 5, 6, 7, 8, 9, 10, 11, 12, 13, 14, 15);
        }

        // ---- A fragments + WMMA: 3 weights x 2 N sub-tiles ----
#pragma unroll
        for (int w = 0; w < 3; ++w) {
            const v8h* arow = (const v8h*)(&Ah[w][mi * 16 + l16][0]);
            const v8h ac0 = arow[half];       // K 0..7  (lo) / 8..15  (hi)
            const v8h ac1 = arow[half + 2];   // K 16..23(lo) / 24..31 (hi)
            const v16h afrag = __builtin_shufflevector(
                ac0, ac1, 0, 1, 2, 3, 4, 5, 6, 7, 8, 9, 10, 11, 12, 13, 14, 15);
#pragma unroll
            for (int n = 0; n < 2; ++n) {
                acc[w][n] = __builtin_amdgcn_wmma_f32_16x16x32_f16(
                    false, afrag, false, bfrag[n], (short)0, acc[w][n],
                    false, false);
            }
        }
        // this iteration's LDS reads must retire before next async writes
        asm volatile("s_wait_dscnt 0x0" ::: "memory");
        __syncthreads();
    }

    // ---- store (f16): C/D layout -> row = g + half*8 (+mi*16)
#pragma unroll
    for (int n = 0; n < 2; ++n) {
        const size_t col = (size_t)tt * TILE_N + ni * 32 + n * 16 + l16;
#pragma unroll
        for (int g = 0; g < 8; ++g) {
            const size_t row = (size_t)ot * TILE_M + mi * 16 + g + half * 8;
            const size_t off = ((size_t)b * ODIM + row) * TDIM + col;
            qh[off] = (_Float16)acc[0][n][g];
            kh[off] = (_Float16)acc[1][n][g];
            vh[off] = (_Float16)acc[2][n][g];
        }
    }
}

// ---------------------------------------------------------------------------
__global__ __launch_bounds__(256) void attn_tap_softmax(
    const _Float16* __restrict__ q,
    const _Float16* __restrict__ k,
    const _Float16* __restrict__ v,
    float* __restrict__ out)
{
    const size_t idx = (size_t)blockIdx.x * 256 + threadIdx.x; // B*O*T total
    const int t = (int)(idx & (TDIM - 1));

    const float qv = (float)q[idx];
    const bool lo = (t >= DIL);
    const bool hi = (t < TDIM - DIL);
    // zero-padded 1x1 conv => out-of-range taps are exactly zero (score & value)
    const float k0 = lo ? (float)k[idx - DIL] : 0.0f;
    const float k1 = (float)k[idx];
    const float k2 = hi ? (float)k[idx + DIL] : 0.0f;
    const float v0 = lo ? (float)v[idx - DIL] : 0.0f;
    const float v1 = (float)v[idx];
    const float v2 = hi ? (float)v[idx + DIL] : 0.0f;

    const float s0 = qv * k0;
    const float s1 = qv * k1;
    const float s2 = qv * k2;
    const float m  = fmaxf(s0, fmaxf(s1, s2));
    const float e0 = __expf(s0 - m);
    const float e1 = __expf(s1 - m);
    const float e2 = __expf(s2 - m);
    out[idx] = (e0 * v0 + e1 * v1 + e2 * v2) / (e0 + e1 + e2);
}

// ---------------------------------------------------------------------------
extern "C" void kernel_launch(void* const* d_in, const int* in_sizes, int n_in,
                              void* d_out, int out_size, void* d_ws, size_t ws_size,
                              hipStream_t stream)
{
    (void)in_sizes; (void)n_in; (void)out_size; (void)ws_size;

    const float* x  = (const float*)d_in[0];
    const float* wq = (const float*)d_in[1];
    const float* wk = (const float*)d_in[2];
    const float* wv = (const float*)d_in[3];

    const size_t BOT = (size_t)BSZ * ODIM * TDIM;  // 16,777,216
    const size_t BCT = (size_t)BSZ * CDIM * TDIM;  // 16,777,216
    const size_t WSZ = (size_t)ODIM * CDIM;        // 262,144 per weight

    _Float16* qh = (_Float16*)d_ws;
    _Float16* kh = qh + BOT;
    _Float16* vh = kh + BOT;
    _Float16* xh = vh + BOT;
    _Float16* wh = xh + BCT;                       // [3][O][C]

    // Phase 0: fp32 -> f16
    cvt_f32_f16<<<(unsigned)(BCT / 4 / 256), 256, 0, stream>>>(x, xh);
    cvt_f32_f16<<<(unsigned)(WSZ / 4 / 256), 256, 0, stream>>>(wq, wh);
    cvt_f32_f16<<<(unsigned)(WSZ / 4 / 256), 256, 0, stream>>>(wk, wh + WSZ);
    cvt_f32_f16<<<(unsigned)(WSZ / 4 / 256), 256, 0, stream>>>(wv, wh + 2 * WSZ);

    // Phase 1: fused QKV GEMM
    dim3 grid(TDIM / TILE_N, ODIM / TILE_M, BSZ);  // (32, 8, 16)
    qkv_gemm_f16_wmma<<<grid, 256, 0, stream>>>(xh, wh, qh, kh, vh);

    // Phase 2: epilogue
    attn_tap_softmax<<<(unsigned)(BOT / 256), 256, 0, stream>>>(qh, kh, vh,
                                                               (float*)d_out);
}
